// GNNLayer_5669356832319
// MI455X (gfx1250) — matrix operand order repaired and checked by reading, not compiled, not run on hardware
//
#include <hip/hip_runtime.h>
#include <math.h>

// ---------------------------------------------------------------------------
// Problem constants (match reference)
// ---------------------------------------------------------------------------
#define N_NODES 100000
#define N_EDGES 3200000
#define IN_F    512
#define OUT_F   256

#define KTILES  (IN_F / 32)    // 16
#define NTILES  (OUT_F / 16)   // 16

// ---------------------------------------------------------------------------
// WMMA vector types (gfx1250, wave32)
// ---------------------------------------------------------------------------
typedef __attribute__((ext_vector_type(16))) __bf16    v16bf;
typedef __attribute__((ext_vector_type(2)))  __bf16    v2bf;
typedef __attribute__((ext_vector_type(8)))  float     v8f;
typedef __attribute__((ext_vector_type(8)))  unsigned  v8u;
typedef __attribute__((ext_vector_type(4)))  unsigned  v4u;

union FragBF {
    v8u   u;    // 8 dwords = 16 packed bf16
    v16bf b;
    v4u   q[2]; // two 16B halves for b128 loads
};

// fp32 pair -> packed bf16 dword (lowers to v_cvt_pk_bf16_f32)
__device__ __forceinline__ unsigned pack2bf(float lo, float hi) {
    v2bf p;
    p.x = (__bf16)lo;
    p.y = (__bf16)hi;
    return __builtin_bit_cast(unsigned, p);
}

// ---------------------------------------------------------------------------
// Kernel 0 (one-shot, tiny): swizzle W [IN_F, OUT_F] f32 into bf16
// B-fragment order so the GEMM loads each fragment as 2x global_load_b128.
//   Wf dword index = ((ktile*NTILES + ntile)*32 + lane)*8 + j
//   value         = pack2bf(W[k][n], W[k+1][n]),
//   k = ktile*32 + (lane>>4)*16 + 2j,  n = ntile*16 + (lane&15)
// Total: 16*16*32*8 dwords = 256 KB (L2 resident).
// ---------------------------------------------------------------------------
__global__ __launch_bounds__(256)
void gnn_wswizzle_kernel(const float* __restrict__ W, unsigned* __restrict__ Wf) {
    const int id    = blockIdx.x * 256 + threadIdx.x;  // 0 .. 65535
    const int j     = id & 7;
    const int lane  = (id >> 3) & 31;
    const int ntile = (id >> 8) & (NTILES - 1);
    const int ktile = id >> 12;

    const int g  = lane >> 4;
    const int n  = ntile * 16 + (lane & 15);
    const int k  = ktile * 32 + g * 16 + 2 * j;

    Wf[id] = pack2bf(W[(size_t)k * OUT_F + n],
                     W[(size_t)(k + 1) * OUT_F + n]);
}

// ---------------------------------------------------------------------------
// Kernel 1: support = tanh(features @ W)
// Block = 128 threads (4 waves); block owns a 16-row stripe and the full
// 256-wide output (wave w -> N-tiles 4w..4w+3).
// Double-buffered A tile in LDS (1 barrier / k-step). Per k-step:
//   - A fragment: 2x ds_load_b128 (16B-aligned, pitch 20 -> conflict-free)
//   - all 4 B fragments loaded up front (one clause of 8 global_load_b128)
//   - 4x v_wmma_f32_16x16x32_bf16 back-to-back
//   - next A tile: float4 load issued before the WMMAs (overlaps compute),
//     cvt + ds_store placed after them
// ---------------------------------------------------------------------------
#define LDS_PITCH 20
#define LDS_TILE  (16 * LDS_PITCH)   // dwords per buffer

__global__ __launch_bounds__(128)
void gnn_gemm_tanh_kernel(const float* __restrict__ A,
                          const unsigned* __restrict__ Wf,
                          float* __restrict__ S) {
    __shared__ __align__(16) unsigned ldsA[2 * LDS_TILE];

    const int tid  = threadIdx.x;
    const int lane = tid & 31;
    const int wave = tid >> 5;           // 0..3
    const int g    = lane >> 4;          // lane half
    const int nl   = lane & 15;

    const int row0   = blockIdx.x * 16;
    const int ntile0 = wave * 4;

    // A-tile cooperative load: 8 threads per row, float4 each
    const int lr = tid >> 3;
    const int lc = (tid & 7) * 4;
    const float*    aptr = A + (size_t)(row0 + lr) * IN_F + lc;
    const unsigned  sidx = lr * LDS_PITCH + (lc >> 1);   // staging dword index
    const unsigned  fidx = nl * LDS_PITCH + g * 8;       // fragment dword index

    v8f acc0 = {}, acc1 = {}, acc2 = {}, acc3 = {};

    // ---- preload tile 0 into buffer 0 ----
    {
        const float4 f = *(const float4*)(aptr);
        ldsA[sidx    ] = pack2bf(f.x, f.y);
        ldsA[sidx + 1] = pack2bf(f.z, f.w);
    }

    for (int ktile = 0; ktile < KTILES; ++ktile) {
        __syncthreads();   // tile `ktile` visible; prev reads of buf^1 done

        const unsigned* buf = ldsA + (ktile & 1) * LDS_TILE;

        // ---- A fragment: two ds_load_b128 ----
        FragBF a;
        a.q[0] = *(const v4u*)(buf + fidx);      // j = 0..3
        a.q[1] = *(const v4u*)(buf + fidx + 4);  // j = 4..7

        // ---- all 4 B fragments up front: one clause of 8 b128 loads ----
        const unsigned* wbase =
            Wf + (((size_t)ktile * NTILES + ntile0) * 32 + lane) * 8;
        FragBF b0, b1, b2, b3;
        b0.q[0] = *(const v4u*)(wbase       );  b0.q[1] = *(const v4u*)(wbase +    4);
        b1.q[0] = *(const v4u*)(wbase + 256 );  b1.q[1] = *(const v4u*)(wbase +  260);
        b2.q[0] = *(const v4u*)(wbase + 512 );  b2.q[1] = *(const v4u*)(wbase +  516);
        b3.q[0] = *(const v4u*)(wbase + 768 );  b3.q[1] = *(const v4u*)(wbase +  772);

        // ---- issue next A-tile global load before the WMMAs ----
        float4 fn;
        const bool have_next = (ktile + 1) < KTILES;
        if (have_next)
            fn = *(const float4*)(aptr + (ktile + 1) * 32);

        // ---- 4 back-to-back WMMAs on the matrix pipe ----
        acc0 = __builtin_amdgcn_wmma_f32_16x16x32_bf16(
                 false, a.b, false, b0.b, (short)0, acc0, false, false);
        acc1 = __builtin_amdgcn_wmma_f32_16x16x32_bf16(
                 false, a.b, false, b1.b, (short)0, acc1, false, false);
        acc2 = __builtin_amdgcn_wmma_f32_16x16x32_bf16(
                 false, a.b, false, b2.b, (short)0, acc2, false, false);
        acc3 = __builtin_amdgcn_wmma_f32_16x16x32_bf16(
                 false, a.b, false, b3.b, (short)0, acc3, false, false);

        // ---- stage next tile into the other buffer (after WMMA issue) ----
        if (have_next) {
            unsigned* nbuf = (unsigned*)ldsA + ((ktile + 1) & 1) * LDS_TILE;
            nbuf[sidx    ] = pack2bf(fn.x, fn.y);
            nbuf[sidx + 1] = pack2bf(fn.z, fn.w);
        }
    }

    // ---- epilogue: tanh + store. C layout: VGPR i -> M = i + g*8, N = nl ----
#pragma unroll
    for (int i = 0; i < 8; ++i) {
        const size_t m = (size_t)(row0 + i + g * 8);
        float* srow = S + m * OUT_F + nl;
        srow[(ntile0 + 0) * 16] = tanhf(acc0[i]);
        srow[(ntile0 + 1) * 16] = tanhf(acc1[i]);
        srow[(ntile0 + 2) * 16] = tanhf(acc2[i]);
        srow[(ntile0 + 3) * 16] = tanhf(acc3[i]);
    }
}

// ---------------------------------------------------------------------------
// Kernel 2: COO SpMM scatter:  dst[row[e]] += val[e] * src[col[e]]
// One wave32 per edge; lane handles 8 features with 128B-coalesced loads
// and global_atomic_add_f32 scatters. src/dst (102.4 MB each) are L2
// resident (192 MB L2), so gathers + atomics stay on-chip.
// ---------------------------------------------------------------------------
#define EDGES_PER_BLOCK 8   // 256 threads = 8 waves

__global__ __launch_bounds__(256)
void gnn_spmm_kernel(const int* __restrict__ erow,
                     const int* __restrict__ ecol,
                     const float* __restrict__ eval,
                     const float* __restrict__ src,
                     float* __restrict__ dst,
                     int n_edges) {
    const int lane = threadIdx.x & 31;
    const int wave = threadIdx.x >> 5;
    const int e    = blockIdx.x * EDGES_PER_BLOCK + wave;
    if (e >= n_edges) return;

    const int   r = erow[e];
    const int   c = ecol[e];
    const float v = eval[e];

    const float* __restrict__ s = src + (size_t)c * OUT_F;
    float*       __restrict__ d = dst + (size_t)r * OUT_F;

#pragma unroll
    for (int j = 0; j < OUT_F / 32; ++j) {
        const int f = j * 32 + lane;       // 32 lanes * 4B = 128B coalesced
        atomicAdd(d + f, v * s[f]);
    }
}

// ---------------------------------------------------------------------------
// kernel_launch
//   d_in : [features, weight, edge_row, edge_col, edge_val]
//   d_out: [output (N*OUT_F) | az (N*OUT_F)] f32
//   d_ws : [support (N*OUT_F f32) | Wf (64K dwords bf16-swizzled W)]
// ---------------------------------------------------------------------------
extern "C" void kernel_launch(void* const* d_in, const int* in_sizes, int n_in,
                              void* d_out, int out_size, void* d_ws, size_t ws_size,
                              hipStream_t stream) {
    const float* features = (const float*)d_in[0];
    const float* weight   = (const float*)d_in[1];
    const int*   edge_row = (const int*)d_in[2];
    const int*   edge_col = (const int*)d_in[3];
    const float* edge_val = (const float*)d_in[4];

    float* out     = (float*)d_out;
    float* az      = out + (size_t)N_NODES * OUT_F;
    float* support = (float*)d_ws;
    unsigned* Wf   = (unsigned*)((char*)d_ws
                      + (size_t)N_NODES * OUT_F * sizeof(float)); // 16B-aligned

    // zero both accumulation targets (graph-capturable memset node)
    hipMemsetAsync(d_out, 0, (size_t)out_size * sizeof(float), stream);

    // one-shot: W -> bf16 fragment-order table (256 KB)
    gnn_wswizzle_kernel<<<256, 256, 0, stream>>>(weight, Wf);

    // support = tanh(features @ W)  -- WMMA bf16 path
    gnn_gemm_tanh_kernel<<<N_NODES / 16, 128, 0, stream>>>(features, Wf, support);

    // output = A_coo @ support ; az = A_coo @ output
    const int nblk = (N_EDGES + EDGES_PER_BLOCK - 1) / EDGES_PER_BLOCK;
    gnn_spmm_kernel<<<nblk, 256, 0, stream>>>(edge_row, edge_col, edge_val,
                                              support, out, N_EDGES);
    gnn_spmm_kernel<<<nblk, 256, 0, stream>>>(edge_row, edge_col, edge_val,
                                              out, az, N_EDGES);
}
